// ETNNEnhancedHierarchicalModel_88622355186346
// MI455X (gfx1250) — matrix operand order repaired and checked by reading, not compiled, not run on hardware
//
#include <hip/hip_runtime.h>
#include <hip/hip_bf16.h>

// ---------------------------------------------------------------------------
// ETNN hierarchical GNN forward for MI455X (gfx1250, wave32, WMMA).
// Dense matmuls -> v_wmma_f32_16x16x32_bf16 (f32 accumulate), weights
// pre-packed to bf16 in LDS-fragment order and staged per k-slice with
// async global->LDS copies (ASYNCcnt). Segment ops -> global f32 atomics.
// ---------------------------------------------------------------------------

typedef __attribute__((ext_vector_type(16))) __bf16 v16bf;
typedef __attribute__((ext_vector_type(8)))  float  v8f;

#define HD 128

#if __has_builtin(__builtin_amdgcn_global_load_async_to_lds_b128)
#define HAVE_ASYNC_LDS 1
#endif

__device__ __forceinline__ unsigned short bf16bits(float f) {
  union { float f; unsigned u; } x; x.f = f;
  unsigned r = x.u + 0x7FFFu + ((x.u >> 16) & 1u);   // round-to-nearest-even
  return (unsigned short)(r >> 16);
}
__device__ __forceinline__ __bf16 f2bf(float f) {
  union { unsigned short s; __bf16 b; } y; y.s = bf16bits(f);
  return y.b;
}

__device__ __forceinline__ float wred_sum(float v) {
  for (int off = 16; off > 0; off >>= 1) v += __shfl_xor(v, off, 32);
  return v;
}

__device__ __forceinline__ void atomicMaxF(float* addr, float val) {
  int* ia = (int*)addr;
  int old = __float_as_int(*addr);
  while (__int_as_float(old) < val) {
    int assumed = old;
    old = atomicCAS(ia, assumed, __float_as_int(val));
    if (old == assumed) break;
  }
}

// Pack fp32 weight [K][128] into bf16 fragment order:
// out[s*4096 + n*32 + kq] = bf16(W[(32s+kq)*128 + n]),  8KB per k-slice.
__global__ void pack_w_kernel(const float* __restrict__ W, unsigned short* __restrict__ out, int K) {
  int idx = blockIdx.x * blockDim.x + threadIdx.x;
  if (idx >= K * HD) return;
  int s  = idx >> 12;
  int n  = (idx >> 5) & 127;
  int kq = idx & 31;
  out[idx] = bf16bits(W[(long)(s * 32 + kq) * HD + n]);
}

// ---------------------------------------------------------------------------
// WMMA GEMM: out[M,128] = act( A[M,K] @ W[K,128] + bias + res )
// Wp = pre-packed bf16 weights (fragment order). A fp32 -> bf16 in-register.
// If A1 != nullptr logical A = concat([A0,A1],axis=1), per-half stride lda.
// Block = 256 thr = 8 waves; 128x128 output tile; wave w -> rows 16w..16w+15.
// ---------------------------------------------------------------------------
__global__ __launch_bounds__(256) void gemm_wmma_kernel(
    const float* __restrict__ A0, const float* __restrict__ A1,
    const unsigned short* __restrict__ Wp, const float* __restrict__ bias,
    const float* __restrict__ res, float* __restrict__ out,
    int M, int K, int lda, int relu)
{
  __shared__ __bf16 ldsB[HD * 32];          // one k-slice, [n][k] = n*32+k
  const int tid   = threadIdx.x;
  const int lane  = tid & 31;
  const int wv    = tid >> 5;
  const int lm    = lane & 15;
  const int khalf = lane >> 4;
  const long rowbase = (long)blockIdx.x * 128 + wv * 16;
  const long mrow = rowbase + lm;
  const bool valid = (mrow < M);

  v8f acc[8];
  for (int t = 0; t < 8; ++t)
    for (int v = 0; v < 8; ++v) acc[t][v] = 0.f;

  for (int kk = 0; kk < K; kk += 32) {
    __syncthreads();                         // previous tile fully consumed
    const unsigned short* wslice = Wp + (kk >> 5) * 4096;
#ifdef HAVE_ASYNC_LDS
    {
      // async byte-copy of the 8KB slice: 256 thr x 2 x 16B, ASYNCcnt tracked
      typedef int v4i __attribute__((vector_size(16)));
      typedef __attribute__((address_space(1))) v4i* gvp;
      typedef __attribute__((address_space(3))) v4i* lvp;
      gvp g = (gvp)((char*)wslice + tid * 16);
      lvp l = (lvp)((char*)ldsB + tid * 16);
      __builtin_amdgcn_global_load_async_to_lds_b128(g, l, 0, 0);
      __builtin_amdgcn_global_load_async_to_lds_b128(g, l, 4096, 0);
#if __has_builtin(__builtin_amdgcn_s_wait_asynccnt)
      __builtin_amdgcn_s_wait_asynccnt(0);
#else
      asm volatile("s_wait_asynccnt 0" ::: "memory");
#endif
    }
#else
    {
      uint4* l = (uint4*)ldsB;
      const uint4* g = (const uint4*)wslice;
      l[tid] = g[tid];
      l[tid + 256] = g[tid + 256];
    }
#endif
    __syncthreads();

    // A fragment: lane row lm; lanes 0-15 K {0-15}, lanes 16-31 K {16-31}
    v16bf a;
    for (int r = 0; r < 2; ++r) {
      int k0 = kk + 16 * khalf + r * 8;
      const float* src = A0; int kx = k0;
      if (A1 != nullptr && k0 >= 128) { src = A1; kx = k0 - 128; }
      const float* p = src + mrow * (long)lda + kx;
      if (valid) {
        if (kk + 32 < K) __builtin_prefetch(p + 32, 0, 1);   // global_prefetch_b8
        float4 u0 = *(const float4*)(p);
        float4 u1 = *(const float4*)(p + 4);
        a[r*8+0] = f2bf(u0.x); a[r*8+1] = f2bf(u0.y); a[r*8+2] = f2bf(u0.z); a[r*8+3] = f2bf(u0.w);
        a[r*8+4] = f2bf(u1.x); a[r*8+5] = f2bf(u1.y); a[r*8+6] = f2bf(u1.z); a[r*8+7] = f2bf(u1.w);
      } else {
        for (int e = 0; e < 8; ++e) a[r*8+e] = f2bf(0.f);
      }
    }

    for (int t = 0; t < 8; ++t) {
      const v16bf* bp = (const v16bf*)&ldsB[(t * 16 + lm) * 32 + khalf * 16];
      acc[t] = __builtin_amdgcn_wmma_f32_16x16x32_bf16(
          false, a, false, *bp, (short)0, acc[t], false, false);
    }
  }

  for (int t = 0; t < 8; ++t) {
    int col = t * 16 + lm;
    float bc = bias ? bias[col] : 0.f;
    for (int v = 0; v < 8; ++v) {
      long row = rowbase + khalf * 8 + v;
      if (row < M) {
        float o = acc[t][v] + bc;
        if (res)  o += res[row * HD + col];
        if (relu) o = fmaxf(o, 0.f);
        out[row * HD + col] = o;
      }
    }
  }
}

// ---------------------------------------------------------------------------
// Support kernels
// ---------------------------------------------------------------------------
__global__ void fill_kernel(float* p, float v, long n) {
  long i = (long)blockIdx.x * blockDim.x + threadIdx.x;
  if (i < n) p[i] = v;
}

__global__ void geo_kernel(const float* __restrict__ pos,
                           const float* __restrict__ Wg1, const float* __restrict__ bg1,
                           const float* __restrict__ Wg2, const float* __restrict__ bg2,
                           float* __restrict__ out, int n)
{
  int i = blockIdx.x; if (i >= n) return;
  __shared__ float hid[64];
  float px = pos[i*3+0], py = pos[i*3+1], pz = pos[i*3+2];
  float sq = px*px + py*py + pz*pz;
  float f0 = sqrtf(sq), f1 = px + py + pz, f2 = sq;
  int tid = threadIdx.x;
  if (tid < 64) {
    float v = f0*Wg1[tid] + f1*Wg1[64+tid] + f2*Wg1[128+tid] + bg1[tid];
    hid[tid] = fmaxf(v, 0.f);
  }
  __syncthreads();
  float o = bg2[tid];
  for (int j = 0; j < 64; ++j) o += hid[j] * Wg2[j*HD + tid];
  out[(long)i*HD + tid] = o;
}

__global__ void node_dot_kernel(const float* __restrict__ h, const float* __restrict__ a,
                                float* __restrict__ out, int n)
{
  int wid = (int)(((long)blockIdx.x * blockDim.x + threadIdx.x) >> 5);
  int lane = threadIdx.x & 31;
  if (wid >= n) return;
  float s = 0.f;
  for (int e = 0; e < 4; ++e) { int c = lane*4 + e; s += h[(long)wid*HD + c] * a[c]; }
  s = wred_sum(s);
  if (lane == 0) out[wid] = s;
}

__global__ void edge_score_kernel(const int* __restrict__ sA, const int* __restrict__ dA,
                                  const float* __restrict__ salo, const float* __restrict__ sahi,
                                  float* __restrict__ score, int E)
{
  int e = blockIdx.x * blockDim.x + threadIdx.x; if (e >= E) return;
  float s = salo[sA[e]] + sahi[dA[e]];
  score[e] = (s > 0.f) ? s : 0.2f * s;          // leaky_relu(0.2)
}

__global__ void seg_max_kernel(const float* __restrict__ score, const int* __restrict__ dA,
                               float* __restrict__ mx, int E)
{
  int e = blockIdx.x * blockDim.x + threadIdx.x; if (e >= E) return;
  atomicMaxF(&mx[dA[e]], score[e]);
}

__global__ void edge_exp_kernel(const float* __restrict__ score, const int* __restrict__ dA,
                                const float* __restrict__ mx, float* __restrict__ w,
                                float* __restrict__ z, int E)
{
  int e = blockIdx.x * blockDim.x + threadIdx.x; if (e >= E) return;
  int d = dA[e];
  float ex = __expf(score[e] - mx[d]);
  w[e] = ex;
  atomicAdd(&z[d], ex);
}

// acc[dst,:] += P[src,:] * (w/(z[dst]+1e-9)); 32 threads/edge, 4 cols each
__global__ void scatter_att_kernel(const float* __restrict__ P, const float* __restrict__ w,
                                   const float* __restrict__ z,
                                   const int* __restrict__ sA, const int* __restrict__ dA,
                                   float* __restrict__ acc, int E)
{
  long idx = (long)blockIdx.x * blockDim.x + threadIdx.x;
  if (idx >= (long)E * 32) return;
  int e = (int)(idx >> 5); int g = ((int)idx & 31) * 4;
  int s = sA[e], d = dA[e];
  float wn = w[e] / (z[d] + 1e-9f);
  float4 pv = *(const float4*)&P[(long)s*HD + g];
  float* ab = &acc[(long)d*HD + g];
  atomicAdd(ab+0, pv.x*wn); atomicAdd(ab+1, pv.y*wn);
  atomicAdd(ab+2, pv.z*wn); atomicAdd(ab+3, pv.w*wn);
}

__global__ void relu_add_kernel(float* __restrict__ out, const float* __restrict__ h,
                                const float* __restrict__ a, const float* __restrict__ b,
                                long n128)
{
  long i = (long)blockIdx.x * blockDim.x + threadIdx.x;
  if (i >= n128) return;
  float v = h[i] + a[i];
  if (b) v += b[i];
  out[i] = fmaxf(v, 0.f);
}

__global__ void edge_d2_kernel(const float* __restrict__ ps, const float* __restrict__ pd,
                               const int* __restrict__ sA, const int* __restrict__ dA,
                               float* __restrict__ d2, int E)
{
  int e = blockIdx.x * blockDim.x + threadIdx.x; if (e >= E) return;
  int s = sA[e], d = dA[e];
  float acc = 0.f;
  for (int k = 0; k < 3; ++k) { float r = pd[d*3+k] - ps[s*3+k]; acc += r*r; }
  d2[e] = acc;
}

// t[e,:] = silu(Ad[dst,:] + As[src,:] + d2[e]*we1last + be1); 32 threads/edge
__global__ void edge_silu_kernel(const float* __restrict__ Ad, const float* __restrict__ As,
                                 const float* __restrict__ d2, const float* __restrict__ wl,
                                 const float* __restrict__ be1,
                                 const int* __restrict__ sA, const int* __restrict__ dA,
                                 float* __restrict__ t, int E)
{
  long idx = (long)blockIdx.x * blockDim.x + threadIdx.x;
  if (idx >= (long)E * 32) return;
  int e = (int)(idx >> 5); int g = ((int)idx & 31) * 4;
  int s = sA[e], d = dA[e];
  float dd = d2[e];
  float4 ad = *(const float4*)&Ad[(long)d*HD + g];
  float4 as = *(const float4*)&As[(long)s*HD + g];
  float4 wv = *(const float4*)&wl[g];
  float4 bv = *(const float4*)&be1[g];
  float4 o;
  float v0 = ad.x + as.x + dd*wv.x + bv.x; o.x = v0 / (1.f + __expf(-v0));
  float v1 = ad.y + as.y + dd*wv.y + bv.y; o.y = v1 / (1.f + __expf(-v1));
  float v2 = ad.z + as.z + dd*wv.z + bv.z; o.z = v2 / (1.f + __expf(-v2));
  float v3 = ad.w + as.w + dd*wv.w + bv.w; o.w = v3 / (1.f + __expf(-v3));
  *(float4*)&t[(long)e*HD + g] = o;
}

__global__ void edge_gate_kernel(const float* __restrict__ m, const float* __restrict__ wg,
                                 float* __restrict__ gate, int E)
{
  int wid = (int)(((long)blockIdx.x * blockDim.x + threadIdx.x) >> 5);
  int lane = threadIdx.x & 31;
  if (wid >= E) return;
  float s = 0.f;
  for (int q = 0; q < 4; ++q) { int c = lane*4 + q; s += m[(long)wid*HD + c] * wg[c]; }
  s = wred_sum(s);
  if (lane == 0) gate[wid] = tanhf(s);
}

__global__ void scatter_feat_kernel(const float* __restrict__ m, const int* __restrict__ dA,
                                    float* __restrict__ facc, int E)
{
  long idx = (long)blockIdx.x * blockDim.x + threadIdx.x;
  if (idx >= (long)E * 32) return;
  int e = (int)(idx >> 5); int g = ((int)idx & 31) * 4;
  int d = dA[e];
  float4 mv = *(const float4*)&m[(long)e*HD + g];
  float* ab = &facc[(long)d*HD + g];
  atomicAdd(ab+0, mv.x); atomicAdd(ab+1, mv.y);
  atomicAdd(ab+2, mv.z); atomicAdd(ab+3, mv.w);
}

__global__ void scatter_pos_kernel(const float* __restrict__ ps, const float* __restrict__ pd,
                                   const float* __restrict__ gate,
                                   const int* __restrict__ sA, const int* __restrict__ dA,
                                   float* __restrict__ dp, float* __restrict__ cnt, int E)
{
  int e = blockIdx.x * blockDim.x + threadIdx.x; if (e >= E) return;
  int s = sA[e], d = dA[e];
  float g = gate[e];
  atomicAdd(&cnt[d], 1.f);
  for (int k = 0; k < 3; ++k) {
    float r = pd[d*3+k] - ps[s*3+k];
    atomicAdd(&dp[d*3+k], r * g);
  }
}

// accum==0: eOut = hB + facc/(cnt+1)   accum==1: eOut += facc/(cnt+1)
__global__ void feat_combine_kernel(float* __restrict__ eOut, const float* __restrict__ hB,
                                    const float* __restrict__ facc, const float* __restrict__ cnt,
                                    long n128, int accum)
{
  long i = (long)blockIdx.x * blockDim.x + threadIdx.x;
  if (i >= n128) return;
  float den = cnt[i >> 7] + 1.f;
  float v = facc[i] / den;
  eOut[i] = accum ? (eOut[i] + v) : (hB[i] + v);
}

// attention fusion + residual + layernorm; one wave per node
__global__ void fuse_kernel(float* __restrict__ h, const float* __restrict__ cb,
                            const float* __restrict__ eb, const float* __restrict__ r,
                            const float* __restrict__ Wf2, const float* __restrict__ bf2,
                            const float* __restrict__ g, const float* __restrict__ b, int n)
{
  int wid = (int)(((long)blockIdx.x * blockDim.x + threadIdx.x) >> 5);
  int lane = threadIdx.x & 31;
  if (wid >= n) return;
  long base = (long)wid * HD;
  float d0 = 0.f, d1 = 0.f;
  for (int q = 0; q < 4; ++q) {
    int c = lane*4 + q; float rv = r[base + c];
    d0 += rv * Wf2[c*2 + 0]; d1 += rv * Wf2[c*2 + 1];
  }
  d0 = wred_sum(d0) + bf2[0];
  d1 = wred_sum(d1) + bf2[1];
  float mx  = fmaxf(d0, d1);
  float ex0 = __expf(d0 - mx), ex1 = __expf(d1 - mx);
  float aw0 = ex0 / (ex0 + ex1), aw1 = ex1 / (ex0 + ex1);
  float xv[4], xs = 0.f, xs2 = 0.f;
  for (int q = 0; q < 4; ++q) {
    int c = lane*4 + q;
    float fv = aw0 * cb[base + c] + aw1 * eb[base + c];
    float x  = fv + h[base + c];
    xv[q] = x; xs += x; xs2 += x*x;
  }
  xs  = wred_sum(xs);
  xs2 = wred_sum(xs2);
  float mu  = xs * (1.f/128.f);
  float var = xs2 * (1.f/128.f) - mu*mu;
  float rs  = rsqrtf(var + 1e-5f);
  for (int q = 0; q < 4; ++q) {
    int c = lane*4 + q;
    h[base + c] = (xv[q] - mu) * rs * g[c] + b[c];
  }
}

__global__ void pos_update_kernel(float* __restrict__ pos, const float* __restrict__ dp,
                                  const float* __restrict__ cnt, int n)
{
  int i = blockIdx.x * blockDim.x + threadIdx.x;
  if (i >= n*3) return;
  pos[i] += dp[i] / (cnt[i/3] + 1.f);
}

__global__ void head_kernel(const float* __restrict__ h,
                            const float* __restrict__ W1, const float* __restrict__ b1,
                            const float* __restrict__ W2, const float* __restrict__ b2,
                            float* __restrict__ out, int n)
{
  int wid = (int)(((long)blockIdx.x * blockDim.x + threadIdx.x) >> 5);
  int lane = threadIdx.x & 31;
  if (wid >= n) return;
  long base = (long)wid * HD;
  float acc = 0.f;
  for (int half = 0; half < 2; ++half) {
    int j = lane + 32*half;
    float hj = b1[j];
    for (int c = 0; c < HD; ++c) hj += h[base + c] * W1[c*64 + j];
    acc += fmaxf(hj, 0.f) * W2[j];
  }
  acc = wred_sum(acc);
  if (lane == 0) out[wid] = acc + b2[0];
}

// ---------------------------------------------------------------------------
// Host orchestration
// ---------------------------------------------------------------------------
extern "C" void kernel_launch(void* const* d_in, const int* in_sizes, int n_in,
                              void* d_out, int out_size, void* d_ws, size_t ws_size,
                              hipStream_t stream)
{
  (void)in_sizes; (void)n_in; (void)out_size; (void)ws_size;
  constexpr int N0 = 200000, N1 = 40000, N2 = 8000;
  constexpr int E01 = 400000, E12 = 80000;

  const float* x    = (const float*)d_in[0];
  const float* pos0 = (const float*)d_in[1];
  const float* pos1 = (const float*)d_in[2];
  const float* pos2 = (const float*)d_in[3];
  const int* e01s = (const int*)d_in[4];
  const int* e01d = (const int*)d_in[5];
  const int* e12s = (const int*)d_in[6];
  const int* e12d = (const int*)d_in[7];

  int pi = 8;
  auto F = [&](int i){ return (const float*)d_in[i]; };
  const float *Wi = F(pi++), *bi = F(pi++);
  const float *Wg1 = F(pi++), *bg1 = F(pi++), *Wg2 = F(pi++), *bg2 = F(pi++);
  struct LayerP { const float *W_up,*a_up,*W_dn,*a_dn,*We1,*be1,*We2,*be2,*wg,*Wf1,*bf1,*Wf2,*bf2,*g,*b; } Lp[2];
  for (int l = 0; l < 2; ++l) {
    Lp[l].W_up=F(pi++); Lp[l].a_up=F(pi++); Lp[l].W_dn=F(pi++); Lp[l].a_dn=F(pi++);
    Lp[l].We1=F(pi++);  Lp[l].be1=F(pi++);  Lp[l].We2=F(pi++);  Lp[l].be2=F(pi++);
    Lp[l].wg=F(pi++);   Lp[l].Wf1=F(pi++);  Lp[l].bf1=F(pi++);  Lp[l].Wf2=F(pi++);
    Lp[l].bf2=F(pi++);  Lp[l].g=F(pi++);    Lp[l].b=F(pi++);
  }
  struct HeadP { const float *W1,*b1,*W2,*b2; } Hd[3];
  for (int k = 0; k < 3; ++k) { Hd[k].W1=F(pi++); Hd[k].b1=F(pi++); Hd[k].W2=F(pi++); Hd[k].b2=F(pi++); }

  // ---- workspace bump allocator (float granularity) ----
  float* ws = (float*)d_ws;
  size_t off = 0;
  auto alloc = [&](size_t n){ float* p = ws + off; off += n; return p; };
  auto allocw = [&](size_t nshort){ return (unsigned short*)alloc((nshort + 1) / 2); };

  float *h0 = alloc((size_t)N0*HD), *h1 = alloc((size_t)N1*HD), *h2 = alloc((size_t)N2*HD);
  float *c0 = alloc((size_t)N0*HD), *c1 = alloc((size_t)N1*HD), *c2 = alloc((size_t)N2*HD);
  float *e0 = alloc((size_t)N0*HD), *e1 = alloc((size_t)N1*HD), *e2 = alloc((size_t)N2*HD);
  float *p0c = alloc((size_t)N0*3), *p1c = alloc((size_t)N1*3), *p2c = alloc((size_t)N2*3);
  float *PA = alloc((size_t)N0*HD), *PB = alloc((size_t)N0*HD);
  float *accA = alloc((size_t)N0*HD), *accB = alloc((size_t)N0*HD);
  float *salo = alloc(N0), *sahi = alloc(N0), *maxb = alloc(N0), *zb = alloc(N0);
  float *cnt0 = alloc(N0), *cnt1a = alloc(N1), *cnt1b = alloc(N1), *cnt2 = alloc(N2);
  float *dp0 = alloc((size_t)N0*3), *dp1a = alloc((size_t)N1*3), *dp1b = alloc((size_t)N1*3), *dp2 = alloc((size_t)N2*3);
  float *sc = alloc(E01), *wgt = alloc(E01), *d2b = alloc(E01), *gate = alloc(E01);
  float *tbuf = alloc((size_t)E01*HD), *mbuf = alloc((size_t)E01*HD);

  // packed bf16 weights (fragment order)
  unsigned short* WiP = allocw((size_t)64*HD);
  struct LayerW { unsigned short *W_up,*W_dn,*We1d,*We1s,*We2,*Wf1; } Lw[2];
  for (int l = 0; l < 2; ++l) {
    Lw[l].W_up = allocw((size_t)HD*HD); Lw[l].W_dn = allocw((size_t)HD*HD);
    Lw[l].We1d = allocw((size_t)HD*HD); Lw[l].We1s = allocw((size_t)HD*HD);
    Lw[l].We2  = allocw((size_t)HD*HD); Lw[l].Wf1  = allocw((size_t)256*HD);
  }

  auto gblk = [](long n, int bs){ return (unsigned)((n + bs - 1) / bs); };
  auto fill = [&](float* p, float v, long n){
    fill_kernel<<<gblk(n,256), 256, 0, stream>>>(p, v, n);
  };
  auto packw = [&](const float* W, unsigned short* out, int K){
    pack_w_kernel<<<gblk((long)K*HD,256), 256, 0, stream>>>(W, out, K);
  };
  auto gemm = [&](const float* A0, const float* A1, const unsigned short* Wp, const float* bias,
                  const float* res, float* out, int M, int K, int lda, int relu){
    gemm_wmma_kernel<<<gblk(M,128), 256, 0, stream>>>(A0, A1, Wp, bias, res, out, M, K, lda, relu);
  };

  // pre-pack all GEMM weights to bf16 fragment order (tiny, once per launch)
  packw(Wi, WiP, 64);
  for (int l = 0; l < 2; ++l) {
    packw(Lp[l].W_up, Lw[l].W_up, HD);
    packw(Lp[l].W_dn, Lw[l].W_dn, HD);
    packw(Lp[l].We1,           Lw[l].We1d, HD);   // We1 rows   0..127 (hd part)
    packw(Lp[l].We1 + 128*HD,  Lw[l].We1s, HD);   // We1 rows 128..255 (hs part)
    packw(Lp[l].We2, Lw[l].We2, HD);
    packw(Lp[l].Wf1, Lw[l].Wf1, 256);
  }

  // private mutable copies of positions (never mutate d_in)
  (void)hipMemcpyAsync(p0c, pos0, (size_t)N0*3*sizeof(float), hipMemcpyDeviceToDevice, stream);
  (void)hipMemcpyAsync(p1c, pos1, (size_t)N1*3*sizeof(float), hipMemcpyDeviceToDevice, stream);
  (void)hipMemcpyAsync(p2c, pos2, (size_t)N2*3*sizeof(float), hipMemcpyDeviceToDevice, stream);

  // ---- input embeddings ----
  geo_kernel<<<N1, 128, 0, stream>>>(p1c, Wg1, bg1, Wg2, bg2, h1, N1);
  geo_kernel<<<N2, 128, 0, stream>>>(p2c, Wg1, bg1, Wg2, bg2, h2, N2);
  geo_kernel<<<N0, 128, 0, stream>>>(p0c, Wg1, bg1, Wg2, bg2, accA, N0);   // geo0 temp
  gemm(x, nullptr, WiP, bi, accA, h0, N0, 64, 64, 0);                       // h0 = x@Wi+bi+geo0

  auto ccmpn = [&](const float* hs, int ns, const float* hd, int nd,
                   const int* sA, const int* dA, int E,
                   const unsigned short* Wp, const float* avec, float* acc){
    gemm(hs, nullptr, Wp, nullptr, nullptr, PA, ns, HD, HD, 0);             // P = hs@W
    node_dot_kernel<<<gblk((long)ns*32,256), 256, 0, stream>>>(PA, avec,    salo, ns);
    node_dot_kernel<<<gblk((long)nd*32,256), 256, 0, stream>>>(hd, avec+HD, sahi, nd);
    fill(maxb, -1e30f, nd); fill(zb, 0.f, nd); fill(acc, 0.f, (long)nd*HD);
    edge_score_kernel<<<gblk(E,256), 256, 0, stream>>>(sA, dA, salo, sahi, sc, E);
    seg_max_kernel  <<<gblk(E,256), 256, 0, stream>>>(sc, dA, maxb, E);
    edge_exp_kernel <<<gblk(E,256), 256, 0, stream>>>(sc, dA, maxb, wgt, zb, E);
    scatter_att_kernel<<<gblk((long)E*32,256), 256, 0, stream>>>(PA, wgt, zb, sA, dA, acc, E);
  };

  auto etnn = [&](const float* hs, int ns, const float* hd, int nd,
                  const float* ps, const float* pd,
                  const int* sA, const int* dA, int E,
                  const LayerP& P, const LayerW& Wl,
                  float* eOut, const float* hBase, int accum, float* dp, float* cnt){
    gemm(hd, nullptr, Wl.We1d, nullptr, nullptr, PA, nd, HD, HD, 0);        // Ad
    gemm(hs, nullptr, Wl.We1s, nullptr, nullptr, PB, ns, HD, HD, 0);        // As
    edge_d2_kernel<<<gblk(E,256), 256, 0, stream>>>(ps, pd, sA, dA, d2b, E);
    edge_silu_kernel<<<gblk((long)E*32,256), 256, 0, stream>>>(
        PA, PB, d2b, P.We1 + 256*HD, P.be1, sA, dA, tbuf, E);
    gemm(tbuf, nullptr, Wl.We2, P.be2, nullptr, mbuf, E, HD, HD, 0);        // m
    edge_gate_kernel<<<gblk((long)E*32,256), 256, 0, stream>>>(mbuf, P.wg, gate, E);
    fill(accA, 0.f, (long)nd*HD); fill(cnt, 0.f, nd); fill(dp, 0.f, (long)nd*3);
    scatter_feat_kernel<<<gblk((long)E*32,256), 256, 0, stream>>>(mbuf, dA, accA, E);
    scatter_pos_kernel <<<gblk(E,256), 256, 0, stream>>>(ps, pd, gate, sA, dA, dp, cnt, E);
    feat_combine_kernel<<<gblk((long)nd*HD,256), 256, 0, stream>>>(
        eOut, hBase, accA, cnt, (long)nd*HD, accum);
  };

  auto fuse = [&](float* h, const float* cb, const float* eb, int n,
                  const LayerP& P, const LayerW& Wl){
    gemm(cb, eb, Wl.Wf1, P.bf1, nullptr, tbuf, n, 256, HD, 1);  // r = relu([c,e]@Wf1+bf1)
    fuse_kernel<<<gblk((long)n*32,256), 256, 0, stream>>>(h, cb, eb, tbuf, P.Wf2, P.bf2, P.g, P.b, n);
  };

  for (int l = 0; l < 2; ++l) {
    const LayerP& P = Lp[l];
    const LayerW& Wl = Lw[l];
    // ---- CCMPN branch ----
    ccmpn(h1, N1, h0, N0, e01d, e01s, E01, Wl.W_dn, P.a_dn, accA);
    relu_add_kernel<<<gblk((long)N0*HD,256), 256, 0, stream>>>(c0, h0, accA, nullptr, (long)N0*HD);
    ccmpn(h0, N0, h1, N1, e01s, e01d, E01, Wl.W_up, P.a_up, accA);
    ccmpn(h2, N2, h1, N1, e12d, e12s, E12, Wl.W_dn, P.a_dn, accB);
    relu_add_kernel<<<gblk((long)N1*HD,256), 256, 0, stream>>>(c1, h1, accA, accB, (long)N1*HD);
    ccmpn(h1, N1, h2, N2, e12s, e12d, E12, Wl.W_up, P.a_up, accA);
    relu_add_kernel<<<gblk((long)N2*HD,256), 256, 0, stream>>>(c2, h2, accA, nullptr, (long)N2*HD);
    // ---- ETNN branch (old positions; updates applied at end of layer) ----
    etnn(h1, N1, h0, N0, p1c, p0c, e01d, e01s, E01, P, Wl, e0, h0, 0, dp0,  cnt0 );
    etnn(h0, N0, h1, N1, p0c, p1c, e01s, e01d, E01, P, Wl, e1, h1, 0, dp1a, cnt1a);
    etnn(h2, N2, h1, N1, p2c, p1c, e12d, e12s, E12, P, Wl, e1, nullptr, 1, dp1b, cnt1b);
    etnn(h1, N1, h2, N2, p1c, p2c, e12s, e12d, E12, P, Wl, e2, h2, 0, dp2,  cnt2 );
    // ---- fusion + residual + LN (h buffers hold the residual originals) ----
    fuse(h0, c0, e0, N0, P, Wl);
    fuse(h1, c1, e1, N1, P, Wl);
    fuse(h2, c2, e2, N2, P, Wl);
    // ---- position updates ----
    pos_update_kernel<<<gblk((long)N0*3,256), 256, 0, stream>>>(p0c, dp0,  cnt0,  N0);
    pos_update_kernel<<<gblk((long)N1*3,256), 256, 0, stream>>>(p1c, dp1a, cnt1a, N1);
    pos_update_kernel<<<gblk((long)N1*3,256), 256, 0, stream>>>(p1c, dp1b, cnt1b, N1);
    pos_update_kernel<<<gblk((long)N2*3,256), 256, 0, stream>>>(p2c, dp2,  cnt2,  N2);
  }

  // ---- output heads ----
  float* out = (float*)d_out;
  head_kernel<<<gblk((long)N0*32,256), 256, 0, stream>>>(h0, Hd[0].W1, Hd[0].b1, Hd[0].W2, Hd[0].b2, out,           N0);
  head_kernel<<<gblk((long)N1*32,256), 256, 0, stream>>>(h1, Hd[1].W1, Hd[1].b1, Hd[1].W2, Hd[1].b2, out + N0,      N1);
  head_kernel<<<gblk((long)N2*32,256), 256, 0, stream>>>(h2, Hd[2].W1, Hd[2].b1, Hd[2].W2, Hd[2].b2, out + N0 + N1, N2);
}
